// MinGRU_22995254903469
// MI455X (gfx1250) — matrix operand order repaired and checked by reading, not compiled, not run on hardware
//
#include <hip/hip_runtime.h>

typedef __attribute__((ext_vector_type(4)))  float   f32x4;
typedef __attribute__((ext_vector_type(8)))  float   v8f;
typedef __attribute__((ext_vector_type(4)))  __bf16  v4bf;
typedef __attribute__((ext_vector_type(8)))  __bf16  v8bf;
typedef __attribute__((ext_vector_type(16))) __bf16  v16bf;

#define BM 128
#define BN 128
#define BK 32
#define LDSTR 40        // padded LDS row stride (elements); 80B = 16B-aligned rows
#define CH 256          // scan chunk length
#define NCH 16          // T / CH

__device__ __forceinline__ v8f wmma_bf16(v16bf a, v16bf b, v8f c) {
    return __builtin_amdgcn_wmma_f32_16x16x32_bf16(
        /*neg_a=*/false, a, /*neg_b=*/false, b,
        /*c_mod=*/(short)0, c, /*reuse_a=*/false, /*reuse_b=*/false);
}

__device__ __forceinline__ v16bf make_frag(const __bf16* p0, const __bf16* p1) {
    v8bf a = *(const v8bf*)p0;
    v8bf b = *(const v8bf*)p1;
    v16bf r;
#pragma unroll
    for (int i = 0; i < 8; ++i) { r[i] = a[i]; r[i + 8] = b[i]; }
    return r;
}

// ---------------------------------------------------------------------------
// Fused dual-GEMM + elementwise epilogue.
//   k  = A @ Wz^T + bz ;  th = A @ Wh^T + bh
//   z = sigmoid(k); a = 1-z; g = th>=0 ? th+0.5 : sigmoid(th); b = z*g
// Writes a -> a_out (workspace), b -> b_out (d_out).
// Split-precision bf16: x ~= hi + lo ; C += Ah*Bh + Ah*Bl + Al*Bh.
// ---------------------------------------------------------------------------
__global__ __launch_bounds__(256)
void mingru_gemm_ew(const float* __restrict__ A,
                    const float* __restrict__ Wz,
                    const float* __restrict__ Wh,
                    const float* __restrict__ bz,
                    const float* __restrict__ bh,
                    float* __restrict__ a_out,
                    float* __restrict__ b_out,
                    int D)
{
    __shared__ __attribute__((aligned(16))) __bf16 sAh[BM * LDSTR];
    __shared__ __attribute__((aligned(16))) __bf16 sAl[BM * LDSTR];
    __shared__ __attribute__((aligned(16))) __bf16 sBh[2][BN * LDSTR];
    __shared__ __attribute__((aligned(16))) __bf16 sBl[2][BN * LDSTR];

    const int tid  = threadIdx.x;
    const int lane = tid & 31;
    const int wid  = tid >> 5;
    const int wm   = wid & 3;      // wave M index (0..3), 32 rows each
    const int wn   = wid >> 2;     // wave N index (0..1), 64 cols each
    const int lr   = lane & 15;    // lane within half-wave
    const int half = lane >> 4;    // 0 or 1

    const int r0 = blockIdx.y * BM;   // global row (b*T + t)
    const int e0 = blockIdx.x * BN;   // global col (e)

    v8f acck[2][4], acch[2][4];
#pragma unroll
    for (int ms = 0; ms < 2; ++ms)
#pragma unroll
        for (int ns = 0; ns < 4; ++ns) {
            acck[ms][ns] = (v8f)0.0f;
            acch[ms][ns] = (v8f)0.0f;
        }

    for (int k0 = 0; k0 < D; k0 += BK) {
        // ---- load A tile (128x32 f32) -> bf16 hi/lo in LDS ----
#pragma unroll
        for (int j = 0; j < 4; ++j) {
            int idx = tid + j * 256;          // 1024 float4 chunks
            int row = idx >> 3;
            int c4  = (idx & 7) << 2;
            f32x4 v = *(const f32x4*)(A + (size_t)(r0 + row) * D + k0 + c4);
            v4bf hv, lv;
#pragma unroll
            for (int i = 0; i < 4; ++i) {
                float x  = v[i];
                __bf16 h = (__bf16)x;
                __bf16 l = (__bf16)(x - (float)h);
                hv[i] = h; lv[i] = l;
            }
            *(v4bf*)&sAh[row * LDSTR + c4] = hv;
            *(v4bf*)&sAl[row * LDSTR + c4] = lv;
        }
        // ---- load both W tiles (128x32 f32 each) -> bf16 hi/lo in LDS ----
#pragma unroll
        for (int mat = 0; mat < 2; ++mat) {
            const float* W = mat ? Wh : Wz;
#pragma unroll
            for (int j = 0; j < 4; ++j) {
                int idx = tid + j * 256;
                int row = idx >> 3;
                int c4  = (idx & 7) << 2;
                f32x4 v = *(const f32x4*)(W + (size_t)(e0 + row) * D + k0 + c4);
                v4bf hv, lv;
#pragma unroll
                for (int i = 0; i < 4; ++i) {
                    float x  = v[i];
                    __bf16 h = (__bf16)x;
                    __bf16 l = (__bf16)(x - (float)h);
                    hv[i] = h; lv[i] = l;
                }
                *(v4bf*)&sBh[mat][row * LDSTR + c4] = hv;
                *(v4bf*)&sBl[mat][row * LDSTR + c4] = lv;
            }
        }
        __syncthreads();

        // ---- A fragments (hi/lo) for the wave's two M-subtiles ----
        // lane<16: K = [0..7] ++ [16..23]; lane>=16: K = [8..15] ++ [24..31]
        v16bf afh[2], afl[2];
#pragma unroll
        for (int ms = 0; ms < 2; ++ms) {
            int arow = wm * 32 + ms * 16 + lr;
            const __bf16* ph = &sAh[arow * LDSTR + half * 8];
            const __bf16* pl = &sAl[arow * LDSTR + half * 8];
            afh[ms] = make_frag(ph, ph + 16);
            afl[ms] = make_frag(pl, pl + 16);
        }

        // ---- B fragments + WMMA ----
        // lane<16: K = [0..7] ++ [16..23]; lane>=16: K = [8..15] ++ [24..31]
#pragma unroll
        for (int mat = 0; mat < 2; ++mat) {
#pragma unroll
            for (int ns = 0; ns < 4; ++ns) {
                int brow = wn * 64 + ns * 16 + lr;
                const __bf16* ph = &sBh[mat][brow * LDSTR + half * 8];
                const __bf16* pl = &sBl[mat][brow * LDSTR + half * 8];
                v16bf bfh = make_frag(ph, ph + 16);
                v16bf bfl = make_frag(pl, pl + 16);
#pragma unroll
                for (int ms = 0; ms < 2; ++ms) {
                    v8f acc = mat ? acch[ms][ns] : acck[ms][ns];
                    acc = wmma_bf16(afh[ms], bfh, acc);   // hi*hi
                    acc = wmma_bf16(afh[ms], bfl, acc);   // hi*lo
                    acc = wmma_bf16(afl[ms], bfh, acc);   // lo*hi
                    if (mat) acch[ms][ns] = acc; else acck[ms][ns] = acc;
                }
            }
        }
        __syncthreads();
    }

    // ---- epilogue: bias + gates, write a (ws) and b (d_out) ----
#pragma unroll
    for (int ns = 0; ns < 4; ++ns) {
        int e = e0 + wn * 64 + ns * 16 + lr;
        float bze = bz[e];
        float bhe = bh[e];
#pragma unroll
        for (int ms = 0; ms < 2; ++ms) {
            int rbase = r0 + wm * 32 + ms * 16 + half * 8;
#pragma unroll
            for (int v = 0; v < 8; ++v) {
                int tr = rbase + v;
                float kk = acck[ms][ns][v] + bze;
                float th = acch[ms][ns][v] + bhe;
                // stable z = sigmoid(kk), a = sigmoid(-kk)
                float ek  = __expf(-fabsf(kk));
                float inv = 1.0f / (1.0f + ek);
                float z = (kk >= 0.0f) ? inv : ek * inv;
                float a = (kk >= 0.0f) ? ek * inv : inv;
                // g(th) = th>=0 ? th+0.5 : sigmoid(th)
                float et = __expf(-fabsf(th));
                float gt = (th >= 0.0f) ? (th + 0.5f) : et / (1.0f + et);
                size_t o = (size_t)tr * D + e;
                a_out[o] = a;
                b_out[o] = z * gt;
            }
        }
    }
}

// ---------------------------------------------------------------------------
// Pass 1: per-chunk carries. For chunk of length CH:
//   P = prod(a), Y = scan result with h_in = 0.
// tid layout: [batch][chunk][e], e fastest.
// ---------------------------------------------------------------------------
__global__ __launch_bounds__(256)
void mingru_scan_carry(const float* __restrict__ a_arr,
                       const float* __restrict__ b_arr,
                       float* __restrict__ P,
                       float* __restrict__ Y,
                       int D, int T)
{
    int tid = blockIdx.x * blockDim.x + threadIdx.x;
    int e   = tid % D;
    int ch  = (tid / D) % NCH;
    int bat = tid / (D * NCH);
    size_t base = ((size_t)bat * T + (size_t)ch * CH) * D + e;
    float p = 1.0f, y = 0.0f;
#pragma unroll 4
    for (int t = 0; t < CH; ++t) {
        float av = a_arr[base];
        float bv = b_arr[base];
        y = fmaf(av, y, bv);
        p *= av;
        base += D;
    }
    P[tid] = p;
    Y[tid] = y;
}

// ---------------------------------------------------------------------------
// Pass 2: sequential combine over the NCH chunks of each (batch, e) channel;
// stores the h entering each chunk.
// ---------------------------------------------------------------------------
__global__ __launch_bounds__(256)
void mingru_scan_combine(const float* __restrict__ P,
                         const float* __restrict__ Y,
                         float* __restrict__ Hin,
                         int D)
{
    int tid = blockIdx.x * blockDim.x + threadIdx.x;   // bat*D + e
    int e   = tid % D;
    int bat = tid / D;
    float hin = 0.0f;
#pragma unroll
    for (int ch = 0; ch < NCH; ++ch) {
        size_t ci = ((size_t)bat * NCH + ch) * D + e;
        Hin[ci] = hin;
        hin = fmaf(P[ci], hin, Y[ci]);
    }
}

// ---------------------------------------------------------------------------
// Pass 3: apply the recurrence with correct h_in, writing h over b in d_out.
// ---------------------------------------------------------------------------
__global__ __launch_bounds__(256)
void mingru_scan_apply(const float* __restrict__ a_arr,
                       const float* __restrict__ Hin,
                       float* __restrict__ out,   // holds b, overwritten with h
                       int D, int T)
{
    int tid = blockIdx.x * blockDim.x + threadIdx.x;
    int e   = tid % D;
    int ch  = (tid / D) % NCH;
    int bat = tid / (D * NCH);
    size_t base = ((size_t)bat * T + (size_t)ch * CH) * D + e;
    float h = Hin[tid];
#pragma unroll 4
    for (int t = 0; t < CH; ++t) {
        float av = a_arr[base];
        float bv = out[base];
        h = fmaf(av, h, bv);
        out[base] = h;
        base += D;
    }
}

extern "C" void kernel_launch(void* const* d_in, const int* in_sizes, int n_in,
                              void* d_out, int out_size, void* d_ws, size_t ws_size,
                              hipStream_t stream)
{
    const float* A  = (const float*)d_in[0];   // (B, T, D)
    const float* Wz = (const float*)d_in[1];   // (D, D)
    const float* bz = (const float*)d_in[2];   // (D,)
    const float* Wh = (const float*)d_in[3];   // (D, D)
    const float* bh = (const float*)d_in[4];   // (D,)
    float* out = (float*)d_out;

    const int D = in_sizes[2];                         // 1024
    const long long total = (long long)in_sizes[0];    // B*T*D
    const int M = (int)(total / D);                    // B*T = 32768
    const int T = CH * NCH;                            // 4096
    const int B = M / T;                               // 8

    // workspace layout: a-array | P | Y | Hin
    float* a_arr = (float*)d_ws;
    float* P   = a_arr + total;
    float* Y   = P + (size_t)B * NCH * D;
    float* Hin = Y + (size_t)B * NCH * D;

    dim3 gg(D / BN, M / BM);                 // (8, 256): N-fastest for A reuse in L2
    mingru_gemm_ew<<<gg, 256, 0, stream>>>(A, Wz, Wh, bz, bh, a_arr, out, D);

    const int nCarry = B * NCH * D;          // 131072
    mingru_scan_carry<<<nCarry / 256, 256, 0, stream>>>(a_arr, out, P, Y, D, T);
    mingru_scan_combine<<<(B * D) / 256, 256, 0, stream>>>(P, Y, Hin, D);
    mingru_scan_apply<<<nCarry / 256, 256, 0, stream>>>(a_arr, Hin, out, D, T);
}
